// ExtendedKANTransformer_4922032521494
// MI455X (gfx1250) — compile-verified, matrix operand order checked by reading, not run on hardware
//
#include <hip/hip_runtime.h>
#include <cmath>

// ---- model dims (match reference) ------------------------------------------
#define B_    64
#define S_    100
#define D_    1024
#define H_    16
#define HD_   64
#define L_    6
#define FF_   4096
#define CL_   6
#define MH_   512
#define M_OUT 30
#define C_OUT 1000
#define NT_   (B_ * S_)          // 6400 token rows

typedef unsigned short u16;
typedef unsigned int   u32;

// CDNA5 WMMA types: 16x16x32 bf16 -> f32
typedef __attribute__((ext_vector_type(16))) __bf16 v16bf;
typedef __attribute__((ext_vector_type(8)))  float  v8f;

union FragU { u16 u[16]; uint4 q[2]; v16bf v; };

// Async global->LDS path (CDNA5): gate on builtin availability so the file
// always compiles; fallback is the proven sync global_load+ds_store path.
#if defined(__gfx1250__) &&                                        \
    __has_builtin(__builtin_amdgcn_global_load_async_to_lds_b128) && \
    __has_builtin(__builtin_amdgcn_s_wait_asynccnt)
#define GEMM_ASYNC 1
typedef int v4i_t __attribute__((ext_vector_type(4)));
typedef __attribute__((address_space(1))) v4i_t* glb_v4i;
typedef __attribute__((address_space(3))) v4i_t* lds_v4i;
#else
#define GEMM_ASYNC 0
#endif

__device__ __forceinline__ u16 f2bf(float f) {            // round-to-nearest-even
    u32 u = __float_as_uint(f);
    u += 0x7FFFu + ((u >> 16) & 1u);
    return (u16)(u >> 16);
}
__device__ __forceinline__ float gelu_exact(float x) {
    return 0.5f * x * (1.0f + erff(x * 0.70710678118654752f));
}

// ============================================================================
// Generic GEMM:  out[m,n] = act( sum_k A[m,k]*W[n,k] + bias[n] ) * scale + res
// A, W are bf16 (row-major, K contiguous, K % 32 == 0). Accumulate f32 in WMMA.
// Block tile 64(M) x 128(N); 8 waves, each wave 32x32 via 2x2 16x16 fragments.
// Async build: double-buffered LDS, GLOBAL_LOAD_ASYNC_TO_LDS_B128, one barrier
// per K-step. Fallback: single-buffer sync staging (two barriers per K-step).
// ============================================================================
__global__ __launch_bounds__(256) void k_gemm(
    const u16* __restrict__ A, const u16* __restrict__ W,
    const float* __restrict__ bias, const float* __restrict__ res,
    float* __restrict__ outF, u16* __restrict__ outB,
    int Mdim, int Ndim, int K, int act, float scale)
{
#if GEMM_ASYNC
    __shared__ u16 As[2 * 64 * 40];    // ping-pong 64x32 tiles, stride 40
    __shared__ u16 Ws[2 * 128 * 40];   // ping-pong 128x32 tiles
#else
    __shared__ u16 As[64 * 40];
    __shared__ u16 Ws[128 * 40];
#endif

    const int m0 = blockIdx.y * 64;
    const int n0 = blockIdx.x * 128;
    const int t  = threadIdx.x;
    const int wave = t >> 5, lane = t & 31;
    const int wm = wave >> 2, wn = wave & 3;           // 2 x 4 wave grid
    const int r16 = lane & 15, khalf = lane >> 4;

    v8f acc[2][2];
    for (int i = 0; i < 2; i++)
        for (int j = 0; j < 2; j++)
            for (int e = 0; e < 8; e++) acc[i][j][e] = 0.0f;

    const int ar = t >> 2, ak = (t & 3) * 8;           // A tile: 8 el / thread
    const int wr = t >> 1, wk = (t & 1) * 16;          // W tile: 16 el / thread

    // fragments: elements 0..7 -> K = khalf*8 + e ; 8..15 -> K = 16 + khalf*8 + e
    auto do_mma = [&](const u16* Ab, const u16* Wb) {
        FragU fa[2], fb[2];
        for (int fm = 0; fm < 2; fm++) {
            int row = wm * 32 + fm * 16 + r16;
            fa[fm].q[0] = *(const uint4*)&Ab[row * 40 + khalf * 8];
            fa[fm].q[1] = *(const uint4*)&Ab[row * 40 + 16 + khalf * 8];
        }
        for (int fn = 0; fn < 2; fn++) {
            int row = wn * 32 + fn * 16 + r16;
            fb[fn].q[0] = *(const uint4*)&Wb[row * 40 + khalf * 8];
            fb[fn].q[1] = *(const uint4*)&Wb[row * 40 + 16 + khalf * 8];
        }
        for (int fm = 0; fm < 2; fm++)
            for (int fn = 0; fn < 2; fn++)
                acc[fm][fn] = __builtin_amdgcn_wmma_f32_16x16x32_bf16(
                    false, fa[fm].v, false, fb[fn].v, (short)0, acc[fm][fn],
                    false, false);
    };

#if GEMM_ASYNC
    auto issue_tile = [&](int kk, int bsel) {
        u16* Ad = &As[bsel * (64 * 40)  + ar * 40 + ak];
        u16* Wd = &Ws[bsel * (128 * 40) + wr * 40 + wk];
        if (m0 + ar < Mdim) {
            __builtin_amdgcn_global_load_async_to_lds_b128(
                (glb_v4i)(A + (size_t)(m0 + ar) * K + kk + ak),
                (lds_v4i)Ad, 0, 0);
        } else {
            *(uint4*)Ad = make_uint4(0, 0, 0, 0);
        }
        if (n0 + wr < Ndim) {
            const u16* wp = W + (size_t)(n0 + wr) * K + kk + wk;
            __builtin_amdgcn_global_load_async_to_lds_b128(
                (glb_v4i)wp, (lds_v4i)Wd, 0, 0);
            __builtin_amdgcn_global_load_async_to_lds_b128(
                (glb_v4i)(wp + 8), (lds_v4i)(Wd + 8), 0, 0);
        } else {
            *(uint4*)Wd       = make_uint4(0, 0, 0, 0);
            *(uint4*)(Wd + 8) = make_uint4(0, 0, 0, 0);
        }
    };

    int buf = 0;
    issue_tile(0, 0);
    for (int k0 = 0; k0 < K; k0 += 32) {
        __builtin_amdgcn_s_wait_asynccnt(0);   // my tile-k loads landed in LDS
        __syncthreads();                       // everyone's tile-k data visible
        if (k0 + 32 < K) issue_tile(k0 + 32, buf ^ 1);  // prefetch next tile
        do_mma(&As[buf * (64 * 40)], &Ws[buf * (128 * 40)]);
        buf ^= 1;
    }
#else
    for (int k0 = 0; k0 < K; k0 += 32) {
        uint4 av = make_uint4(0, 0, 0, 0);
        if (m0 + ar < Mdim)
            av = *(const uint4*)(A + (size_t)(m0 + ar) * K + k0 + ak);
        *(uint4*)&As[ar * 40 + ak] = av;

        uint4 w0 = make_uint4(0, 0, 0, 0), w1 = make_uint4(0, 0, 0, 0);
        if (n0 + wr < Ndim) {
            const u16* wp = W + (size_t)(n0 + wr) * K + k0 + wk;
            w0 = *(const uint4*)(wp);
            w1 = *(const uint4*)(wp + 8);
        }
        *(uint4*)&Ws[wr * 40 + wk]     = w0;
        *(uint4*)&Ws[wr * 40 + wk + 8] = w1;

        if (k0 + 32 < K)  // prefetch next A tile -> global_prefetch_b8
            __builtin_prefetch(A + (size_t)(m0 + ar) * K + (k0 + 32) + ak, 0, 1);
        __syncthreads();
        do_mma(As, Ws);
        __syncthreads();
    }
#endif

    // epilogue: C/D layout: lanes 0-15 M=v, lanes 16-31 M=v+8; N = lane%16
    for (int fm = 0; fm < 2; fm++)
        for (int fn = 0; fn < 2; fn++)
            for (int v = 0; v < 8; v++) {
                int mg = m0 + wm * 32 + fm * 16 + v + 8 * khalf;
                int ng = n0 + wn * 32 + fn * 16 + r16;
                if (mg < Mdim && ng < Ndim) {
                    float rv = acc[fm][fn][v];
                    if (bias) rv += bias[ng];
                    if (act == 1)      rv = gelu_exact(rv);
                    else if (act == 2) rv = fmaxf(rv, 0.0f);
                    rv *= scale;
                    size_t idx = (size_t)mg * Ndim + ng;
                    if (res)  rv += res[idx];
                    if (outF) outF[idx] = rv;
                    if (outB) outB[idx] = f2bf(rv);
                }
            }
}

// ============================================================================
// Fused attention for one (s,h): scores = Q K^T / 8 over batch axis (64x64),
// softmax, O = P V. Q/K/V/P staged as bf16 in LDS, WMMA for both GEMMs.
// ============================================================================
__global__ __launch_bounds__(256) void k_attn(
    const u16* __restrict__ qkv,   // [B*S, 3D] bf16
    u16* __restrict__ ao)          // [B*S, D]  bf16
{
    __shared__ u16  Qs[64 * 72], Ks[64 * 72], Vs[64 * 72], Ps[64 * 72];
    __shared__ float Sc[64 * 68];

    const int s = blockIdx.x / H_, h = blockIdx.x % H_;
    const int t = threadIdx.x, wave = t >> 5, lane = t & 31;
    const int r16 = lane & 15, khalf = lane >> 4;

    {   // load Q,K,V 64x64 bf16 tiles (16 el / thread each)
        int r = t >> 2, c = (t & 3) * 16;
        size_t base = ((size_t)r * S_ + s) * (3 * D_) + (size_t)h * HD_ + c;
        *(uint4*)&Qs[r * 72 + c]     = *(const uint4*)(qkv + base);
        *(uint4*)&Qs[r * 72 + c + 8] = *(const uint4*)(qkv + base + 8);
        *(uint4*)&Ks[r * 72 + c]     = *(const uint4*)(qkv + base + D_);
        *(uint4*)&Ks[r * 72 + c + 8] = *(const uint4*)(qkv + base + D_ + 8);
        *(uint4*)&Vs[r * 72 + c]     = *(const uint4*)(qkv + base + 2 * D_);
        *(uint4*)&Vs[r * 72 + c + 8] = *(const uint4*)(qkv + base + 2 * D_ + 8);
    }
    __syncthreads();

    // scores: 4x4 grid of 16x16 tiles, each wave owns 2 tiles
    for (int tt = 0; tt < 2; tt++) {
        int tile = wave * 2 + tt, tm = tile >> 2, tn = tile & 3;
        v8f acc; for (int e = 0; e < 8; e++) acc[e] = 0.0f;
        for (int k0 = 0; k0 < HD_; k0 += 32) {
            FragU fa, fb;
            int ra = tm * 16 + r16, rb = tn * 16 + r16;
            fa.q[0] = *(const uint4*)&Qs[ra * 72 + k0 + khalf * 8];
            fa.q[1] = *(const uint4*)&Qs[ra * 72 + k0 + 16 + khalf * 8];
            fb.q[0] = *(const uint4*)&Ks[rb * 72 + k0 + khalf * 8];
            fb.q[1] = *(const uint4*)&Ks[rb * 72 + k0 + 16 + khalf * 8];
            acc = __builtin_amdgcn_wmma_f32_16x16x32_bf16(
                false, fa.v, false, fb.v, (short)0, acc, false, false);
        }
        for (int v = 0; v < 8; v++) {
            int ml = tm * 16 + v + 8 * khalf, nl = tn * 16 + r16;
            Sc[ml * 68 + nl] = acc[v] * 0.125f;       // 1/sqrt(HD)
        }
    }
    __syncthreads();

    if (t < 64) {  // f32 softmax, one row per thread
        float mx = -3.4e38f;
        for (int j = 0; j < 64; j++) mx = fmaxf(mx, Sc[t * 68 + j]);
        float sum = 0.0f;
        for (int j = 0; j < 64; j++) {
            float e = __expf(Sc[t * 68 + j] - mx);
            Sc[t * 68 + j] = e; sum += e;
        }
        float inv = 1.0f / sum;
        for (int j = 0; j < 64; j++) Ps[t * 72 + j] = f2bf(Sc[t * 68 + j] * inv);
    }
    __syncthreads();

    // O = P @ V  (V read transposed from LDS for the B fragment)
    for (int tt = 0; tt < 2; tt++) {
        int tile = wave * 2 + tt, tm = tile >> 2, tn = tile & 3;
        v8f acc; for (int e = 0; e < 8; e++) acc[e] = 0.0f;
        for (int k0 = 0; k0 < 64; k0 += 32) {
            FragU fa, fb;
            int ra = tm * 16 + r16;
            fa.q[0] = *(const uint4*)&Ps[ra * 72 + k0 + khalf * 8];
            fa.q[1] = *(const uint4*)&Ps[ra * 72 + k0 + 16 + khalf * 8];
            int col = tn * 16 + r16;
            for (int e = 0; e < 16; e++) {
                int kk = (e < 8) ? (khalf * 8 + e) : (16 + khalf * 8 + (e - 8));
                fb.u[e] = Vs[(k0 + kk) * 72 + col];
            }
            acc = __builtin_amdgcn_wmma_f32_16x16x32_bf16(
                false, fa.v, false, fb.v, (short)0, acc, false, false);
        }
        for (int v = 0; v < 8; v++) {
            int bi = tm * 16 + v + 8 * khalf;   // batch (l) index
            int dl = tn * 16 + r16;             // head-dim index
            ao[((size_t)bi * S_ + s) * D_ + (size_t)h * HD_ + dl] = f2bf(acc[v]);
        }
    }
}

// ============================================================================
// LayerNorm over last dim (D=1024), one row per block (256 thr x 4 el)
// ============================================================================
__global__ __launch_bounds__(256) void k_layernorm(
    const float* __restrict__ x, const float* __restrict__ g,
    const float* __restrict__ b, u16* __restrict__ outB,
    float* __restrict__ outF)
{
    __shared__ float red[256];
    const int row = blockIdx.x;
    const float* xr = x + (size_t)row * D_;
    float v[4]; float s = 0.0f;
    for (int i = 0; i < 4; i++) { v[i] = xr[threadIdx.x + i * 256]; s += v[i]; }
    red[threadIdx.x] = s; __syncthreads();
    for (int off = 128; off > 0; off >>= 1) {
        if (threadIdx.x < off) red[threadIdx.x] += red[threadIdx.x + off];
        __syncthreads();
    }
    float mu = red[0] * (1.0f / D_);
    __syncthreads();
    float s2 = 0.0f;
    for (int i = 0; i < 4; i++) { float d = v[i] - mu; s2 += d * d; }
    red[threadIdx.x] = s2; __syncthreads();
    for (int off = 128; off > 0; off >>= 1) {
        if (threadIdx.x < off) red[threadIdx.x] += red[threadIdx.x + off];
        __syncthreads();
    }
    float rstd = rsqrtf(red[0] * (1.0f / D_) + 1e-5f);
    for (int i = 0; i < 4; i++) {
        int c = threadIdx.x + i * 256;
        float o = (v[i] - mu) * rstd * g[c] + b[c];
        size_t idx = (size_t)row * D_ + c;
        if (outB) outB[idx] = f2bf(o);
        if (outF) outF[idx] = o;
    }
}

// ---- small elementwise kernels ---------------------------------------------
__global__ void k_cvt(const float* __restrict__ src, u16* __restrict__ dst, size_t n) {
    for (size_t i = (size_t)blockIdx.x * blockDim.x + threadIdx.x; i < n;
         i += (size_t)gridDim.x * blockDim.x)
        dst[i] = f2bf(src[i]);
}
__global__ void k_addpos(float* __restrict__ x, const float* __restrict__ pos) {
    size_t n = (size_t)NT_ * D_;
    for (size_t i = (size_t)blockIdx.x * blockDim.x + threadIdx.x; i < n;
         i += (size_t)gridDim.x * blockDim.x) {
        int d = (int)(i % D_);
        int s = (int)((i / D_) % S_);
        x[i] += pos[(size_t)s * D_ + d];
    }
}
__global__ void k_spike(float* __restrict__ x, u16* __restrict__ xb, size_t n) {
    for (size_t i = (size_t)blockIdx.x * blockDim.x + threadIdx.x; i < n;
         i += (size_t)gridDim.x * blockDim.x) {
        float v = x[i];
        float a = (fabsf(v) > 0.55f) ? ((v > 0.0f) ? 1.0f : -1.0f) : 0.0f;
        x[i] = a; xb[i] = f2bf(a);
    }
}
__global__ void k_meanS(const float* __restrict__ in, u16* __restrict__ outB) {
    size_t n = (size_t)B_ * D_;
    for (size_t i = (size_t)blockIdx.x * blockDim.x + threadIdx.x; i < n;
         i += (size_t)gridDim.x * blockDim.x) {
        int b = (int)(i / D_), d = (int)(i % D_);
        float s = 0.0f;
        for (int ss = 0; ss < S_; ss++) s += in[((size_t)b * S_ + ss) * D_ + d];
        outB[i] = f2bf(s * (1.0f / S_));
    }
}

// ============================================================================
// Host orchestration (graph-capture safe: only kernel launches on `stream`)
// ============================================================================
static inline void gemm(hipStream_t st, const u16* A, const u16* W,
                        const float* bias, const float* res,
                        float* outF, u16* outB,
                        int M, int N, int K, int act, float scale) {
    dim3 g((N + 127) / 128, (M + 63) / 64);
    k_gemm<<<g, 256, 0, st>>>(A, W, bias, res, outF, outB, M, N, K, act, scale);
}
static inline void cvt(hipStream_t st, const float* s, u16* d, size_t n) {
    int blocks = (int)((n + 1023) / 1024); if (blocks > 4096) blocks = 4096;
    k_cvt<<<blocks, 256, 0, st>>>(s, d, n);
}

extern "C" void kernel_launch(void* const* d_in, const int* in_sizes, int n_in,
                              void* d_out, int out_size, void* d_ws, size_t ws_size,
                              hipStream_t stream) {
    const float* x      = (const float*)d_in[0];
    const float* emb_w  = (const float*)d_in[1];
    const float* emb_b  = (const float*)d_in[2];
    const float* pos    = (const float*)d_in[3];
    const float* ln1_g  = (const float*)d_in[4];
    const float* ln1_b  = (const float*)d_in[5];
    const float* qkv_w  = (const float*)d_in[6];
    const float* qkv_b  = (const float*)d_in[7];
    const float* out_w  = (const float*)d_in[8];
    const float* out_b  = (const float*)d_in[9];
    const float* ln2_g  = (const float*)d_in[10];
    const float* ln2_b  = (const float*)d_in[11];
    const float* mlp_w1 = (const float*)d_in[12];
    const float* mlp_b1 = (const float*)d_in[13];
    const float* mlp_w2 = (const float*)d_in[14];
    const float* mlp_b2 = (const float*)d_in[15];
    const float* cort_w = (const float*)d_in[16];
    const float* cort_b = (const float*)d_in[17];
    const float* mot_w1 = (const float*)d_in[18];
    const float* mot_b1 = (const float*)d_in[19];
    const float* mot_w2 = (const float*)d_in[20];
    const float* mot_b2 = (const float*)d_in[21];
    const float* norm_g = (const float*)d_in[22];
    const float* norm_b = (const float*)d_in[23];
    const float* head_w = (const float*)d_in[24];
    const float* head_b = (const float*)d_in[25];

    float* logits = (float*)d_out;               // [B, C]
    float* motor  = (float*)d_out + (size_t)B_ * C_OUT;  // [B, M]

    // bump allocator over workspace
    char* ws = (char*)d_ws; size_t off = 0;
    auto alloc = [&](size_t bytes) -> void* {
        void* p = ws + off; off = (off + bytes + 255) & ~(size_t)255; return p;
    };

    // bf16 weight mirrors
    u16* wb_emb  = (u16*)alloc((size_t)D_ * D_ * 2);
    u16* wb_qkv  = (u16*)alloc((size_t)L_ * 3 * D_ * D_ * 2);
    u16* wb_out  = (u16*)alloc((size_t)L_ * D_ * D_ * 2);
    u16* wb_m1   = (u16*)alloc((size_t)L_ * FF_ * D_ * 2);
    u16* wb_m2   = (u16*)alloc((size_t)L_ * D_ * FF_ * 2);
    u16* wb_cort = (u16*)alloc((size_t)CL_ * D_ * D_ * 2);
    u16* wb_mot1 = (u16*)alloc((size_t)MH_ * D_ * 2);
    u16* wb_mot2 = (u16*)alloc((size_t)M_OUT * MH_ * 2);
    u16* wb_head = (u16*)alloc((size_t)C_OUT * D_ * 2);

    // activations
    float* xf    = (float*)alloc((size_t)NT_ * D_ * 4);   // f32 residual stream
    u16*   xb    = (u16*)  alloc((size_t)NT_ * D_ * 2);   // bf16 copies
    u16*   hb    = (u16*)  alloc((size_t)NT_ * D_ * 2);
    u16*   qkvb  = (u16*)  alloc((size_t)NT_ * 3 * D_ * 2);
    u16*   aob   = (u16*)  alloc((size_t)NT_ * D_ * 2);
    u16*   tb    = (u16*)  alloc((size_t)NT_ * FF_ * 2);  // mlp hidden
    float* cf    = (float*)alloc((size_t)NT_ * D_ * 4);   // cortical f32
    u16*   cb1   = (u16*)  alloc((size_t)NT_ * D_ * 2);
    u16*   cb2   = (u16*)  alloc((size_t)NT_ * D_ * 2);
    float* lnf   = (float*)alloc((size_t)NT_ * D_ * 4);   // LN(c) f32
    u16*   poolb = (u16*)  alloc((size_t)B_ * D_ * 2);
    u16*   mhb   = (u16*)  alloc((size_t)B_ * MH_ * 2);
    u16*   xob   = (u16*)  alloc((size_t)B_ * D_ * 2);

    // ---- weight conversion -------------------------------------------------
    cvt(stream, emb_w,  wb_emb,  (size_t)D_ * D_);
    cvt(stream, qkv_w,  wb_qkv,  (size_t)L_ * 3 * D_ * D_);
    cvt(stream, out_w,  wb_out,  (size_t)L_ * D_ * D_);
    cvt(stream, mlp_w1, wb_m1,   (size_t)L_ * FF_ * D_);
    cvt(stream, mlp_w2, wb_m2,   (size_t)L_ * D_ * FF_);
    cvt(stream, cort_w, wb_cort, (size_t)CL_ * D_ * D_);
    cvt(stream, mot_w1, wb_mot1, (size_t)MH_ * D_);
    cvt(stream, mot_w2, wb_mot2, (size_t)M_OUT * MH_);
    cvt(stream, head_w, wb_head, (size_t)C_OUT * D_);
    cvt(stream, x,      xb,      (size_t)NT_ * D_);

    // ---- embedding + positional --------------------------------------------
    gemm(stream, xb, wb_emb, emb_b, nullptr, xf, nullptr, NT_, D_, D_, 0, 1.0f);
    k_addpos<<<4096, 256, 0, stream>>>(xf, pos);

    // ---- transformer layers ------------------------------------------------
    for (int i = 0; i < L_; i++) {
        k_layernorm<<<NT_, 256, 0, stream>>>(xf, ln1_g + (size_t)i * D_,
                                             ln1_b + (size_t)i * D_, hb, nullptr);
        gemm(stream, hb, wb_qkv + (size_t)i * 3 * D_ * D_,
             qkv_b + (size_t)i * 3 * D_, nullptr, nullptr, qkvb,
             NT_, 3 * D_, D_, 0, 1.0f);
        k_attn<<<S_ * H_, 256, 0, stream>>>(qkvb, aob);
        gemm(stream, aob, wb_out + (size_t)i * D_ * D_,
             out_b + (size_t)i * D_, xf, xf, nullptr, NT_, D_, D_, 0, 1.0f);
        k_layernorm<<<NT_, 256, 0, stream>>>(xf, ln2_g + (size_t)i * D_,
                                             ln2_b + (size_t)i * D_, hb, nullptr);
        gemm(stream, hb, wb_m1 + (size_t)i * FF_ * D_,
             mlp_b1 + (size_t)i * FF_, nullptr, nullptr, tb,
             NT_, FF_, D_, 1, 1.0f);                       // GELU
        gemm(stream, tb, wb_m2 + (size_t)i * D_ * FF_,
             mlp_b2 + (size_t)i * D_, xf, xf, nullptr,
             NT_, D_, FF_, 0, 1.0f);                       // METABOLIC = 1.0
        k_spike<<<4096, 256, 0, stream>>>(xf, xb, (size_t)NT_ * D_);
    }

    // ---- cortical stack (dense + exact GELU) -------------------------------
    const u16* cin = xb;
    for (int j = 0; j < CL_; j++) {
        u16* cout = (j & 1) ? cb2 : cb1;
        gemm(stream, cin, wb_cort + (size_t)j * D_ * D_,
             cort_b + (size_t)j * D_, nullptr, cf, cout, NT_, D_, D_, 1, 1.0f);
        cin = cout;
    }

    // ---- motor head: relu(pool @ w1^T + b1) @ w2^T + b2, scaled by ACH -----
    k_meanS<<<256, 256, 0, stream>>>(cf, poolb);
    gemm(stream, poolb, wb_mot1, mot_b1, nullptr, nullptr, mhb,
         B_, MH_, D_, 2, 1.0f);                            // ReLU
    gemm(stream, mhb, wb_mot2, mot_b2, nullptr, motor, nullptr,
         B_, M_OUT, MH_, 0, 0.5f);                         // * ACH

    // ---- classification head: LN(c).mean(S) @ head_w^T + head_b -----------
    k_layernorm<<<NT_, 256, 0, stream>>>(cf, norm_g, norm_b, nullptr, lnf);
    k_meanS<<<256, 256, 0, stream>>>(lnf, xob);
    gemm(stream, xob, wb_head, head_b, nullptr, logits, nullptr,
         B_, C_OUT, D_, 0, 1.0f);

    (void)in_sizes; (void)n_in; (void)out_size; (void)ws_size;
}
